// PairwiseCosinesLayer_66563403153867
// MI455X (gfx1250) — matrix operand order repaired and checked by reading, not compiled, not run on hardware
//
#include <hip/hip_runtime.h>

// ---------------------------------------------------------------------------
// Pairwise cosine similarity:
//   C[b,n,m] = dot(l2norm(x[b,n,:]+1e-7), l2norm(ref[m,:]))
// == GEMM C = A * B^T with A=Xn [8192x1024], B=Refn [8192x1024], fp32 out.
// Strategy: normalize to f16 in workspace, then WMMA f16->f32 tiled GEMM with
// CDNA5 async global->LDS staging (ASYNCcnt) double-buffered in LDS.
// ---------------------------------------------------------------------------

typedef __attribute__((ext_vector_type(16))) _Float16 v16h;
typedef __attribute__((ext_vector_type(8)))  float    v8f;
typedef _Float16 half_t;

#define D_DIM   1024              // K
#define M_X     8192              // rows of x (4*2048)
#define M_REF   8192              // rows of ref
#define BM      128
#define BN      128
#define BK      32                // one WMMA K-step per LDS tile
#define LDSS    40                // 32 halves + 8 pad (80B row stride: conflict-free)
#define NTHREADS 256

union F16x16 {
    v16h  v;
    uint4 q[2];
};

struct H4 { half_t a, b, c, d; };

// ---------------------------------------------------------------------------
// Row-wise L2 normalize fp32 -> f16.  One block per row of 1024 elements.
// ---------------------------------------------------------------------------
__global__ __launch_bounds__(NTHREADS)
void l2norm_f16_kernel(const float* __restrict__ in, half_t* __restrict__ out,
                       float eps)
{
    __shared__ float red[8];
    const int row  = blockIdx.x;
    const int tid  = threadIdx.x;
    const int lane = tid & 31;
    const int wave = tid >> 5;

    const float4* rin4 = (const float4*)(in + (size_t)row * D_DIM);
    float4 v = rin4[tid];                 // 256 threads * 4 = 1024
    v.x += eps; v.y += eps; v.z += eps; v.w += eps;

    float ss = v.x * v.x + v.y * v.y + v.z * v.z + v.w * v.w;
    #pragma unroll
    for (int off = 16; off > 0; off >>= 1)
        ss += __shfl_xor(ss, off, 32);
    if (lane == 0) red[wave] = ss;
    __syncthreads();
    if (wave == 0) {
        float t = (lane < 8) ? red[lane] : 0.0f;
        #pragma unroll
        for (int off = 4; off > 0; off >>= 1)
            t += __shfl_xor(t, off, 32);
        if (lane == 0) red[0] = t;
    }
    __syncthreads();

    const float scale = rsqrtf(fmaxf(red[0], 1e-12f));
    H4 h;
    h.a = (half_t)(v.x * scale);
    h.b = (half_t)(v.y * scale);
    h.c = (half_t)(v.z * scale);
    h.d = (half_t)(v.w * scale);
    ((H4*)(out + (size_t)row * D_DIM))[tid] = h;
}

// ---------------------------------------------------------------------------
// Tiled WMMA GEMM: C[M_X x M_REF] = A[M_X x K] * B[M_REF x K]^T   (f16 -> f32)
// 256 threads = 8 waves; block tile 128x128; wave tile 64x32 (4x2 WMMA tiles).
// Double-buffered LDS fed by GLOBAL_LOAD_ASYNC_TO_LDS_B128 (ASYNCcnt),
// one s_wait_asynccnt + barrier per K iteration.
// ---------------------------------------------------------------------------
__device__ __forceinline__ void stage_async(unsigned ldsA, unsigned ldsB,
                                            const half_t* ga, const half_t* gb)
{
    unsigned long long a64 = (unsigned long long)(uintptr_t)ga;
    unsigned long long b64 = (unsigned long long)(uintptr_t)gb;
    // Each lane copies 32B of A and 32B of B straight into LDS.
    // INST_OFFSET is added to BOTH the LDS and global addresses (ISA 08 §4.4).
    asm volatile(
        "global_load_async_to_lds_b128 %0, %2, off\n\t"
        "global_load_async_to_lds_b128 %0, %2, off offset:16\n\t"
        "global_load_async_to_lds_b128 %1, %3, off\n\t"
        "global_load_async_to_lds_b128 %1, %3, off offset:16"
        :
        : "v"(ldsA), "v"(ldsB), "v"(a64), "v"(b64)
        : "memory");
}

__global__ __launch_bounds__(NTHREADS)
void cosine_gemm_wmma(const half_t* __restrict__ A, const half_t* __restrict__ B,
                      float* __restrict__ C)
{
    __shared__ half_t As[2][BM * LDSS];   // 2 * 128*40*2B = 20480 B
    __shared__ half_t Bs[2][BN * LDSS];   // 20480 B            (total 40 KB)

    const int tid    = threadIdx.x;
    const int lane   = tid & 31;
    const int wave   = tid >> 5;
    const int waveM  = wave >> 2;         // 0..1  (rows of 64)
    const int waveN  = wave & 3;          // 0..3  (cols of 32)
    const int laneLo = lane & 15;
    const int laneHi = lane >> 4;         // 0/1

    const int blockM = blockIdx.y * BM;
    const int blockN = blockIdx.x * BN;

    // Tile loader: each thread moves 32B of A and 32B of B per K-step.
    const int ldRow = tid >> 1;           // 0..127
    const int ldCol = (tid & 1) * 16;     // 0 or 16 halves

    const half_t* aptr = A + (size_t)(blockM + ldRow) * D_DIM + ldCol;
    const half_t* bptr = B + (size_t)(blockN + ldRow) * D_DIM + ldCol;

    // LDS byte offsets for this thread's staging slot in each buffer
    // (low 32 bits of a generic __shared__ pointer == LDS offset).
    const unsigned ldsA0 = (unsigned)(uintptr_t)&As[0][ldRow * LDSS + ldCol];
    const unsigned ldsA1 = (unsigned)(uintptr_t)&As[1][ldRow * LDSS + ldCol];
    const unsigned ldsB0 = (unsigned)(uintptr_t)&Bs[0][ldRow * LDSS + ldCol];
    const unsigned ldsB1 = (unsigned)(uintptr_t)&Bs[1][ldRow * LDSS + ldCol];

    v8f acc[4][2] = {};

    // Stage tile 0 into buffer 0 and make it visible.
    stage_async(ldsA0, ldsB0, aptr, bptr);
    asm volatile("s_wait_asynccnt 0x0" ::: "memory");
    __syncthreads();

    const int NT = D_DIM / BK;            // 32 K-steps
    int cur = 0;

    for (int kt = 0; kt < NT; ++kt) {
        // Kick off async staging of the next tile into the other buffer;
        // it overlaps the entire WMMA phase below.
        if (kt + 1 < NT)
            stage_async(cur ? ldsA0 : ldsA1, cur ? ldsB0 : ldsB1,
                        aptr + (size_t)(kt + 1) * BK,
                        bptr + (size_t)(kt + 1) * BK);

        // A fragments (16x32 f16): lane = row M; halves 0-7 = K[8*hi+0..7],
        // halves 8-15 = K[16+8*hi+0..7]  (ISA 7.12.2 16-bit A layout)
        v16h afrag[4];
        #pragma unroll
        for (int tm = 0; tm < 4; ++tm) {
            const half_t* p = &As[cur][(waveM * 64 + tm * 16 + laneLo) * LDSS];
            F16x16 u;
            u.q[0] = *(const uint4*)(p + laneHi * 8);
            u.q[1] = *(const uint4*)(p + 16 + laneHi * 8);
            afrag[tm] = u.v;
        }
        // B fragments (32x16 f16): lane = col N; halves 0-7 = K[8*hi+0..7],
        // halves 8-15 = K[16+8*hi+0..7]  (same K order as A)
        v16h bfrag[2];
        #pragma unroll
        for (int tn = 0; tn < 2; ++tn) {
            const half_t* p = &Bs[cur][(waveN * 32 + tn * 16 + laneLo) * LDSS];
            F16x16 u;
            u.q[0] = *(const uint4*)(p + laneHi * 8);
            u.q[1] = *(const uint4*)(p + 16 + laneHi * 8);
            bfrag[tn] = u.v;
        }

        #pragma unroll
        for (int tm = 0; tm < 4; ++tm)
            #pragma unroll
            for (int tn = 0; tn < 2; ++tn)
                acc[tm][tn] = __builtin_amdgcn_wmma_f32_16x16x32_f16(
                    /*neg_a=*/false, afrag[tm],
                    /*neg_b=*/false, bfrag[tn],
                    /*c_mod=*/(short)0, acc[tm][tn],
                    /*reuse_a=*/false, /*reuse_b=*/false);

        // Next tile must be fully landed in LDS and visible to all waves.
        asm volatile("s_wait_asynccnt 0x0" ::: "memory");
        __syncthreads();
        cur ^= 1;
    }

    // Epilogue: C/D layout — VGPR r holds (M = r + 8*laneHi, N = laneLo)
    #pragma unroll
    for (int tm = 0; tm < 4; ++tm) {
        const int rowBase = blockM + waveM * 64 + tm * 16 + laneHi * 8;
        #pragma unroll
        for (int tn = 0; tn < 2; ++tn) {
            const int col = blockN + waveN * 32 + tn * 16 + laneLo;
            #pragma unroll
            for (int r = 0; r < 8; ++r)
                C[(size_t)(rowBase + r) * M_REF + col] = acc[tm][tn][r];
        }
    }
}

// ---------------------------------------------------------------------------
extern "C" void kernel_launch(void* const* d_in, const int* in_sizes, int n_in,
                              void* d_out, int out_size, void* d_ws, size_t ws_size,
                              hipStream_t stream)
{
    const float* x   = (const float*)d_in[0];   // [4,2048,1024] fp32
    const float* ref = (const float*)d_in[1];   // [8192,1024]  fp32
    float* out       = (float*)d_out;           // [4,2048,8192] fp32

    half_t* Ah = (half_t*)d_ws;                     // 16 MB
    half_t* Bh = Ah + (size_t)M_X * D_DIM;          // 16 MB

    l2norm_f16_kernel<<<M_X,   NTHREADS, 0, stream>>>(x,   Ah, 1e-7f);
    l2norm_f16_kernel<<<M_REF, NTHREADS, 0, stream>>>(ref, Bh, 0.0f);

    dim3 grid(M_REF / BN, M_X / BM);                // 64 x 64
    cosine_gemm_wmma<<<grid, NTHREADS, 0, stream>>>(Ah, Bh, out);
}
